// NStepLSTM_35407710389082
// MI455X (gfx1250) — matrix operand order repaired and checked by reading, not compiled, not run on hardware
//
#include <hip/hip_runtime.h>

// ---------------------------------------------------------------------------
// Persistent 2-layer LSTM for MI455X (gfx1250, wave32, WMMA bf16).
// Weights are converted once f32->bf16 and pre-swizzled into WMMA B-fragment
// order (L2-resident: 25.8 MB << 192 MB L2). One persistent kernel of 64 WGs
// runs all 127 time steps; cell state lives in registers; h vectors are
// exchanged via double-buffered global bf16 buffers with a grid barrier.
// Weight pointers are laundered per time step (prevents LICM hoist + scratch
// spill) and all fragment loads go through address_space(1) pointers so they
// lower to global_load_b128 (not flat_load, which also burns DScnt).
// ---------------------------------------------------------------------------

#define NWG   64
#define TCOND 96
#define TOUT  127   // 96 teacher-forced + 31 autoregressive preds

typedef __attribute__((ext_vector_type(16))) __bf16 v16bf;
typedef __attribute__((ext_vector_type(8)))  __bf16 v8bf;
typedef __attribute__((ext_vector_type(8)))  float  v8f;

#define GAS __attribute__((address_space(1)))

// ---- workspace layout (all offsets multiples of 256 bytes) ----------------
static constexpr size_t OFF_CNT   = 0;                            // u32 barrier counter
static constexpr size_t OFF_HB    = 256;                          // 4 x [64x1024] bf16 (h0 ping/pong, h1 ping/pong)
static constexpr size_t OFF_XALL  = OFF_HB    + (size_t)262144*2; // [96][64][64] bf16
static constexpr size_t OFF_XCUR  = OFF_XALL  + (size_t)393216*2; // [64][64] bf16 (AR input)
static constexpr size_t OFF_B0    = OFF_XCUR  + 8192;             // bias0 = bih0+bhh0, f32[4096]
static constexpr size_t OFF_B1    = OFF_B0    + 16384;            // bias1, f32[4096]
static constexpr size_t OFF_BOUT  = OFF_B1    + 16384;            // f32[64]
static constexpr size_t OFF_PWIH0 = OFF_BOUT  + 256;              // packed bf16 4096x64
static constexpr size_t OFF_PWHH0 = OFF_PWIH0 + (size_t)262144*2; // packed bf16 4096x1024
static constexpr size_t OFF_PWIH1 = OFF_PWHH0 + (size_t)4194304*2;
static constexpr size_t OFF_PWHH1 = OFF_PWIH1 + (size_t)4194304*2;
static constexpr size_t OFF_PWOUT = OFF_PWHH1 + (size_t)4194304*2; // packed bf16 64x1024
// total ~27.2 MB

__device__ __forceinline__ unsigned short f2bf_bits(float x) {
  union { float f; unsigned int u; } v; v.f = x;
  unsigned int r = v.u + 0x7FFFu + ((v.u >> 16) & 1u);  // round-to-nearest-even
  return (unsigned short)(r >> 16);
}
__device__ __forceinline__ float fast_sigmoid(float x) {
  return 1.0f / (1.0f + __expf(-x));
}
__device__ __forceinline__ float fast_tanh(float x) {
  float xc = fminf(fmaxf(x, -15.0f), 15.0f);
  float e  = __expf(2.0f * xc);
  return (e - 1.0f) / (e + 1.0f);
}

// Opaque pointer pass-through: defeats LICM/CSE across the time loop so the
// weight-fragment loads stay inside the loop (direct L2-resident streams)
// instead of being hoisted and spilled to scratch.
__device__ __forceinline__ const unsigned short* launder(const unsigned short* p) {
  unsigned long long v = (unsigned long long)p;
  asm volatile("" : "+v"(v));
  return (const unsigned short*)v;
}

// Global-address-space loads: force global_load_b128 even for laundered ptrs.
__device__ __forceinline__ v16bf gload32B(const unsigned short* p) {
  return *reinterpret_cast<const GAS v16bf*>((unsigned long long)p);
}
__device__ __forceinline__ v8bf gload16B(const unsigned short* p) {
  return *reinterpret_cast<const GAS v8bf*>((unsigned long long)p);
}

// ---------------------------------------------------------------------------
// Pre-pass: pack weight W [N rows x K cols] (used as B = W^T, KxN) into WMMA
// B-fragment order. Fragment (nt,kt): lane l holds 16 contiguous bf16 so the
// main loop issues one 32B-per-lane load per fragment.
//   B[k][n] = W[n][k];  lane l -> n = nt*16 + (l&15), k = kt*32 + (l&16) + e
// ---------------------------------------------------------------------------
__global__ void pack_weight_kernel(const float* __restrict__ W,
                                   unsigned short* __restrict__ P,
                                   int K, int KT, int total) {
  int tid = blockIdx.x * blockDim.x + threadIdx.x;
  if (tid >= total) return;
  int e    = tid & 15;
  int lane = (tid >> 4) & 31;
  int frag = tid >> 9;
  int kt = frag % KT;
  int nt = frag / KT;
  int n  = nt * 16 + (lane & 15);
  int k  = kt * 32 + (lane & 16) + e;
  P[tid] = f2bf_bits(W[(size_t)n * K + k]);
}

// Pre-pass: combine biases, transpose batch to [t][b][c] bf16, zero h state
// and the grid-barrier counter (re-run every launch => graph-replay safe).
__global__ void prep_kernel(const float* __restrict__ batch,
                            const float* __restrict__ bih0, const float* __restrict__ bhh0,
                            const float* __restrict__ bih1, const float* __restrict__ bhh1,
                            const float* __restrict__ bout,
                            unsigned short* __restrict__ xall,
                            unsigned short* __restrict__ hb,
                            float* __restrict__ bias0, float* __restrict__ bias1,
                            float* __restrict__ boutf,
                            unsigned int* __restrict__ counter) {
  int tid = blockIdx.x * blockDim.x + threadIdx.x;
  if (tid < TCOND * 64 * 64) {
    int t = tid >> 12;
    int r = tid & 4095;
    int b = r >> 6;
    int c = r & 63;
    xall[tid] = f2bf_bits(batch[(size_t)b * (TCOND * 64) + t * 64 + c]);
  }
  if (tid < 262144) hb[tid] = 0;  // all four h buffers
  if (tid < 4096) {
    bias0[tid] = bih0[tid] + bhh0[tid];
    bias1[tid] = bih1[tid] + bhh1[tid];
  }
  if (tid < 64)  boutf[tid] = bout[tid];
  if (tid == 0)  *counter   = 0u;
}

// A-fragment gather from row-major bf16 matrix (contiguous K, matches B pack):
// lane<16 -> K {0..15}, lane>=16 -> K {16..31} (kb pre-offset).
__device__ __forceinline__ v16bf load_afrag(const unsigned short* __restrict__ A,
                                            int rowstride, int row, int kb) {
  const unsigned short* p = A + (size_t)row * rowstride + kb + (kb & 31);
  v8bf lo = gload16B(p);        // k .. k+7
  v8bf hi = gload16B(p + 8);    // k+8 .. k+15
  return __builtin_shufflevector(lo, hi, 0,1,2,3,4,5,6,7,8,9,10,11,12,13,14,15);
}

// Accumulate A[rows]*W^T over KT k-tiles into two 16x16 tiles sharing B.
__device__ __forceinline__ void gemm_tile2(const unsigned short* __restrict__ A, int rowstride,
                                           const unsigned short* __restrict__ P, int KT, int nt,
                                           int row0, int row1, int hiK, int lane,
                                           v8f& acc0, v8f& acc1) {
  const unsigned short* wp = P + ((size_t)nt * KT) * 512 + lane * 16;
  for (int kt = 0; kt < KT; ++kt) {
    v16bf bf = gload32B(wp);
    __builtin_prefetch((const void*)(wp + 4096), 0, 1);  // stream weights L2->WGP$
    wp += 512;
    int kb = kt * 32 + hiK;
    v16bf a0 = load_afrag(A, rowstride, row0, kb);
    v16bf a1 = load_afrag(A, rowstride, row1, kb);
    acc0 = __builtin_amdgcn_wmma_f32_16x16x32_bf16(false, a0, false, bf, (short)0, acc0, false, false);
    acc1 = __builtin_amdgcn_wmma_f32_16x16x32_bf16(false, a1, false, bf, (short)0, acc1, false, false);
  }
}

__device__ __forceinline__ void grid_barrier(unsigned int* counter, unsigned int& target) {
  __syncthreads();
  target += NWG;
  if (threadIdx.x == 0) {
    __threadfence();              // release our global writes
    atomicAdd(counter, 1u);
    while (__hip_atomic_load(counter, __ATOMIC_RELAXED, __HIP_MEMORY_SCOPE_AGENT) < target) {
      __builtin_amdgcn_s_sleep(2);
    }
    __threadfence();              // acquire others' writes
  }
  __syncthreads();
}

// ---------------------------------------------------------------------------
// Persistent kernel: 64 WGs x 256 threads (8 wave32).
// WG wg owns hidden slice [16*wg, 16*wg+16). wave w: gate group gt=w&3,
// M-half mh=w>>2 (two 16-row tiles). Cell state in registers (4 f32/thread
// per layer). 3 grid barriers per step.
// ---------------------------------------------------------------------------
__global__ __launch_bounds__(256, 1)
void lstm_persistent_kernel(const unsigned short* __restrict__ xall,
                            unsigned short* __restrict__ hb,
                            unsigned short* __restrict__ xcur,
                            const unsigned short* __restrict__ pWih0,
                            const unsigned short* __restrict__ pWhh0,
                            const unsigned short* __restrict__ pWih1,
                            const unsigned short* __restrict__ pWhh1,
                            const unsigned short* __restrict__ pWout,
                            const float* __restrict__ bias0,
                            const float* __restrict__ bias1,
                            const float* __restrict__ boutf,
                            float* __restrict__ out,
                            unsigned int* counter) {
  __shared__ float lds_g[4][64][17];  // [gate][batch][hidden-slice], padded

  const int tid  = threadIdx.x;
  const int wg   = blockIdx.x;
  const int wave = tid >> 5;
  const int lane = tid & 31;
  const int gt   = wave & 3;          // gate group (i,f,g,o)
  const int mh   = wave >> 2;         // M half: tiles {2mh, 2mh+1}
  const int lj   = lane & 15;         // column within 16-wide tile
  const int hi8  = (lane & 16) ? 8 : 0;
  const int j0   = wg * 16;           // hidden slice base
  const int nt   = gt * 64 + wg;      // gate column tile (of 256)
  const int r0   = mh * 32 + lj;      // lane's A row, tile 0
  const int r1   = mh * 32 + 16 + lj; // lane's A row, tile 1

  float c0[4] = {0.f, 0.f, 0.f, 0.f};
  float c1[4] = {0.f, 0.f, 0.f, 0.f};
  unsigned int target = 0;

  for (int t = 0; t < TOUT; ++t) {
    const int p = t & 1;  // ping-pong h buffers
    const unsigned short* h0r = hb + p * 65536;
    unsigned short*       h0w = hb + (p ^ 1) * 65536;
    const unsigned short* h1r = hb + 131072 + p * 65536;
    unsigned short*       h1w = hb + 131072 + (p ^ 1) * 65536;
    const unsigned short* xp  = (t < TCOND) ? (xall + t * 4096) : xcur;

    // Re-launder weight bases every step: keeps the fragment loads inside
    // the loop as direct global_load_b128 streams from the packed arrays.
    const unsigned short* wIh0 = launder(pWih0);
    const unsigned short* wHh0 = launder(pWhh0);
    const unsigned short* wIh1 = launder(pWih1);
    const unsigned short* wHh1 = launder(pWhh1);

    // ---------------- layer 0: gates = x*Wih0^T + h0*Whh0^T ----------------
    {
      v8f acc0 = {}; v8f acc1 = {};
      gemm_tile2(xp,   64,  wIh0,  2, nt, r0, r1, hi8, lane, acc0, acc1);
      gemm_tile2(h0r, 1024, wHh0, 32, nt, r0, r1, hi8, lane, acc0, acc1);
#pragma unroll
      for (int r = 0; r < 8; ++r) {
        lds_g[gt][mh * 32 + hi8 + r][lj]      = acc0[r];
        lds_g[gt][mh * 32 + 16 + hi8 + r][lj] = acc1[r];
      }
    }
    __syncthreads();
#pragma unroll
    for (int q = 0; q < 4; ++q) {
      int idx = tid + 256 * q;          // 64 batch x 16 hidden
      int b = idx >> 4, j = idx & 15;
      float gi = lds_g[0][b][j] + bias0[j0 + j];
      float gf = lds_g[1][b][j] + bias0[1024 + j0 + j];
      float gg = lds_g[2][b][j] + bias0[2048 + j0 + j];
      float go = lds_g[3][b][j] + bias0[3072 + j0 + j];
      float c  = fast_sigmoid(gf) * c0[q] + fast_sigmoid(gi) * fast_tanh(gg);
      c0[q] = c;
      float h = fast_sigmoid(go) * fast_tanh(c);
      h0w[b * 1024 + j0 + j] = f2bf_bits(h);
    }
    grid_barrier(counter, target);  // all WGs see new h0

    // ---------------- layer 1: gates = h0*Wih1^T + h1*Whh1^T ---------------
    {
      v8f acc0 = {}; v8f acc1 = {};
      gemm_tile2(h0w, 1024, wIh1, 32, nt, r0, r1, hi8, lane, acc0, acc1);
      gemm_tile2(h1r, 1024, wHh1, 32, nt, r0, r1, hi8, lane, acc0, acc1);
#pragma unroll
      for (int r = 0; r < 8; ++r) {
        lds_g[gt][mh * 32 + hi8 + r][lj]      = acc0[r];
        lds_g[gt][mh * 32 + 16 + hi8 + r][lj] = acc1[r];
      }
    }
    __syncthreads();
#pragma unroll
    for (int q = 0; q < 4; ++q) {
      int idx = tid + 256 * q;
      int b = idx >> 4, j = idx & 15;
      float gi = lds_g[0][b][j] + bias1[j0 + j];
      float gf = lds_g[1][b][j] + bias1[1024 + j0 + j];
      float gg = lds_g[2][b][j] + bias1[2048 + j0 + j];
      float go = lds_g[3][b][j] + bias1[3072 + j0 + j];
      float c  = fast_sigmoid(gf) * c1[q] + fast_sigmoid(gi) * fast_tanh(gg);
      c1[q] = c;
      float h = fast_sigmoid(go) * fast_tanh(c);
      h1w[b * 1024 + j0 + j] = f2bf_bits(h);
    }
    grid_barrier(counter, target);  // all WGs see new h1

    // ---------------- pred = h1*Wout^T + bout (WGs 0..3, waves 0..3) -------
    if (wg < 4 && wave < 4) {
      v8f acc = {};
      const unsigned short* wp = launder(pWout) + ((size_t)wg * 32) * 512 + lane * 16;
      int rr = wave * 16 + lj;
      for (int kt = 0; kt < 32; ++kt) {
        v16bf bf = gload32B(wp);
        wp += 512;
        v16bf a = load_afrag(h1w, 1024, rr, kt * 32 + hi8);
        acc = __builtin_amdgcn_wmma_f32_16x16x32_bf16(false, a, false, bf, (short)0, acc, false, false);
      }
      int c = j0 + lj;            // output channel
      float bo = boutf[c];
#pragma unroll
      for (int r = 0; r < 8; ++r) {
        int b = wave * 16 + hi8 + r;
        float v = acc[r] + bo;
        out[(size_t)b * (TOUT * 64) + (size_t)t * 64 + c] = v;
        xcur[b * 64 + c] = f2bf_bits(v);  // AR feedback input
      }
    }
    grid_barrier(counter, target);  // xcur ready for next step
  }
}

// ---------------------------------------------------------------------------
extern "C" void kernel_launch(void* const* d_in, const int* in_sizes, int n_in,
                              void* d_out, int out_size, void* d_ws, size_t ws_size,
                              hipStream_t stream) {
  const float* batch = (const float*)d_in[0];
  const float* Wih0  = (const float*)d_in[1];
  const float* Whh0  = (const float*)d_in[2];
  const float* bih0  = (const float*)d_in[3];
  const float* bhh0  = (const float*)d_in[4];
  const float* Wih1  = (const float*)d_in[5];
  const float* Whh1  = (const float*)d_in[6];
  const float* bih1  = (const float*)d_in[7];
  const float* bhh1  = (const float*)d_in[8];
  const float* Wout  = (const float*)d_in[9];
  const float* bout  = (const float*)d_in[10];

  char* ws = (char*)d_ws;
  unsigned int*   counter = (unsigned int*)(ws + OFF_CNT);
  unsigned short* hb      = (unsigned short*)(ws + OFF_HB);
  unsigned short* xall    = (unsigned short*)(ws + OFF_XALL);
  unsigned short* xcur    = (unsigned short*)(ws + OFF_XCUR);
  float*          bias0   = (float*)(ws + OFF_B0);
  float*          bias1   = (float*)(ws + OFF_B1);
  float*          boutf   = (float*)(ws + OFF_BOUT);
  unsigned short* pWih0   = (unsigned short*)(ws + OFF_PWIH0);
  unsigned short* pWhh0   = (unsigned short*)(ws + OFF_PWHH0);
  unsigned short* pWih1   = (unsigned short*)(ws + OFF_PWIH1);
  unsigned short* pWhh1   = (unsigned short*)(ws + OFF_PWHH1);
  unsigned short* pWout   = (unsigned short*)(ws + OFF_PWOUT);
  float*          out     = (float*)d_out;

  // pre-pass: biases + batch transpose + state/barrier init
  prep_kernel<<<(TCOND * 64 * 64 + 255) / 256, 256, 0, stream>>>(
      batch, bih0, bhh0, bih1, bhh1, bout, xall, hb, bias0, bias1, boutf, counter);

  // pre-pass: weight f32 -> bf16 fragment packing (once; L2-resident after)
  pack_weight_kernel<<<262144 / 256, 256, 0, stream>>>(Wih0, pWih0, 64, 2, 262144);
  pack_weight_kernel<<<4194304 / 256, 256, 0, stream>>>(Whh0, pWhh0, 1024, 32, 4194304);
  pack_weight_kernel<<<4194304 / 256, 256, 0, stream>>>(Wih1, pWih1, 1024, 32, 4194304);
  pack_weight_kernel<<<4194304 / 256, 256, 0, stream>>>(Whh1, pWhh1, 1024, 32, 4194304);
  pack_weight_kernel<<<65536 / 256, 256, 0, stream>>>(Wout, pWout, 1024, 32, 65536);

  // persistent time loop
  lstm_persistent_kernel<<<NWG, 256, 0, stream>>>(
      xall, hb, xcur, pWih0, pWhh0, pWih1, pWhh1, pWout,
      bias0, bias1, boutf, out, counter);
}
